// ClusterMemory_87411174408305
// MI455X (gfx1250) — matrix operand order repaired and checked by reading, not compiled, not run on hardware
//
#include <hip/hip_runtime.h>

// ---------------------------------------------------------------------------
// CDNA5 (gfx1250) implementation. wave32, WMMA 16x16x32 bf16 with bf16x3
// fp32-emulation split. All heavy matmuls go through v_wmma_f32_16x16x32_bf16.
// ---------------------------------------------------------------------------

typedef __attribute__((ext_vector_type(16))) __bf16 v16bf;
typedef __attribute__((ext_vector_type(8)))  float  v8f;

#define B_DIM 256
#define S_DIM 8192
#define F_DIM 2048
#define K_NB  20

// split an fp32 value array into bf16 hi + bf16 lo (a ~= hi + lo)
__device__ __forceinline__ void split16(const float* __restrict__ f,
                                        v16bf& hi, v16bf& lo) {
#pragma unroll
  for (int j = 0; j < 16; ++j) {
    float v = f[j];
    __bf16 h = (__bf16)v;
    hi[j] = h;
    lo[j] = (__bf16)(v - (float)h);
  }
}

__device__ __forceinline__ void load16(const float* __restrict__ p, float* f) {
  const float4* q = (const float4*)p;
#pragma unroll
  for (int i = 0; i < 4; ++i) {
    float4 t = q[i];
    f[i * 4 + 0] = t.x; f[i * 4 + 1] = t.y;
    f[i * 4 + 2] = t.z; f[i * 4 + 3] = t.w;
  }
}

// C[256, N] = scale * A[256, K] * B[N, K]^T   (both operands K-contiguous, NT)
// grid.x = N/32 (32-col stripe per block); block = 64 threads = 2 waves;
// wave w covers rows [w*128, w*128+128) as 8 M-tiles, and BOTH 16-col N-tiles
// of the stripe (16 v8f accumulators) so each A fragment feeds 6 WMMAs.
__global__ __launch_bounds__(64) void gemm_nt_bf16x3(
    const float* __restrict__ A, const float* __restrict__ B,
    float* __restrict__ C, int K, int lda, int ldb, int ldc, float scale) {
  const int lane  = threadIdx.x & 31;
  const int mBase = (threadIdx.x >> 5) * 128;
  const int nBase = blockIdx.x * 32;
  const int idx16 = lane & 15;   // m-row for A frag, n-col for B/C frags
  const int half  = lane >> 4;

  v8f acc[8][2];
#pragma unroll
  for (int i = 0; i < 8; ++i)
#pragma unroll
    for (int nt = 0; nt < 2; ++nt)
#pragma unroll
      for (int r = 0; r < 8; ++r) acc[i][nt][r] = 0.0f;

  // B fragment source: lane holds col n = idx16, rows k = k0 + half*16 + j
  const float* brow0 = B + (size_t)(nBase + idx16) * ldb + half * 16;
  const float* brow1 = B + (size_t)(nBase + 16 + idx16) * ldb + half * 16;

  for (int k0 = 0; k0 < K; k0 += 32) {
    // prefetch the B stream (features) ahead: lowers to global_prefetch_b8
    __builtin_prefetch(brow0 + k0 + 64, 0, 3);
    __builtin_prefetch(brow1 + k0 + 64, 0, 3);

    float bfv[16];
    v16bf bhi0, blo0, bhi1, blo1;
    load16(brow0 + k0, bfv);
    split16(bfv, bhi0, blo0);
    load16(brow1 + k0, bfv);
    split16(bfv, bhi1, blo1);

#pragma unroll
    for (int mt = 0; mt < 8; ++mt) {
      // A fragment: lane holds row m; e0..15 = A[m, k0+half*16 .. +15]
      const float* arow = A + (size_t)(mBase + mt * 16 + idx16) * lda + k0;
      const float4* ap0 = (const float4*)(arow + half * 16);
      const float4* ap1 = (const float4*)(arow + half * 16 + 8);
      float4 t0 = ap0[0], t1 = ap0[1], t2 = ap1[0], t3 = ap1[1];
      float afv[16] = {t0.x, t0.y, t0.z, t0.w, t1.x, t1.y, t1.z, t1.w,
                       t2.x, t2.y, t2.z, t2.w, t3.x, t3.y, t3.z, t3.w};
      v16bf ahi, alo;
      split16(afv, ahi, alo);
      acc[mt][0] = __builtin_amdgcn_wmma_f32_16x16x32_bf16(
          false, ahi, false, bhi0, (short)0, acc[mt][0], false, false);
      acc[mt][0] = __builtin_amdgcn_wmma_f32_16x16x32_bf16(
          false, ahi, false, blo0, (short)0, acc[mt][0], false, false);
      acc[mt][0] = __builtin_amdgcn_wmma_f32_16x16x32_bf16(
          false, alo, false, bhi0, (short)0, acc[mt][0], false, false);
      acc[mt][1] = __builtin_amdgcn_wmma_f32_16x16x32_bf16(
          false, ahi, false, bhi1, (short)0, acc[mt][1], false, false);
      acc[mt][1] = __builtin_amdgcn_wmma_f32_16x16x32_bf16(
          false, ahi, false, blo1, (short)0, acc[mt][1], false, false);
      acc[mt][1] = __builtin_amdgcn_wmma_f32_16x16x32_bf16(
          false, alo, false, bhi1, (short)0, acc[mt][1], false, false);
    }
  }

  // D layout: lane holds col n = idx16, VGPR r -> row (r + 8*half) of the tile
#pragma unroll
  for (int mt = 0; mt < 8; ++mt)
#pragma unroll
    for (int nt = 0; nt < 2; ++nt)
#pragma unroll
      for (int r = 0; r < 8; ++r) {
        int m = mBase + mt * 16 + r + half * 8;
        C[(size_t)m * ldc + nBase + nt * 16 + idx16] = acc[mt][nt][r] * scale;
      }
}

// x = inputs0 / max(||inputs0||_row, eps); also zeroes the 3 scalar outputs
__global__ __launch_bounds__(256) void normalize_kernel(
    const float* __restrict__ in, float* __restrict__ x, float* __restrict__ out) {
  __shared__ float sred[256];
  int b = blockIdx.x, tid = threadIdx.x;
  if (b == 0 && tid < 3) out[tid] = 0.0f;
  const float* r = in + (size_t)b * F_DIM;
  float v[8], ss = 0.f;
#pragma unroll
  for (int i = 0; i < 8; ++i) { v[i] = r[tid + i * 256]; ss += v[i] * v[i]; }
  sred[tid] = ss; __syncthreads();
  for (int off = 128; off > 0; off >>= 1) {
    if (tid < off) sred[tid] += sred[tid + off];
    __syncthreads();
  }
  float inv = 1.0f / fmaxf(sqrtf(sred[0]), 1e-12f);
  float* xr = x + (size_t)b * F_DIM;
#pragma unroll
  for (int i = 0; i < 8; ++i) xr[tid + i * 256] = v[i] * inv;
}

// per-row max + log-sum-exp for outputs / logits0 / logits1
__global__ __launch_bounds__(256) void row_stats_kernel(
    const float* __restrict__ outputs, const float* __restrict__ l0,
    const float* __restrict__ l1, float* __restrict__ stats) {
  __shared__ float sdata[256];
  int which = blockIdx.x >> 8;      // 0..2
  int row = blockIdx.x & 255;
  const float* src = (which == 0) ? outputs : ((which == 1) ? l0 : l1);
  const float* p = src + (size_t)row * S_DIM;
  int tid = threadIdx.x;
  float m = -3.4e38f;
  for (int s = tid; s < S_DIM; s += 256) m = fmaxf(m, p[s]);
  sdata[tid] = m; __syncthreads();
  for (int off = 128; off > 0; off >>= 1) {
    if (tid < off) sdata[tid] = fmaxf(sdata[tid], sdata[tid + off]);
    __syncthreads();
  }
  m = sdata[0]; __syncthreads();
  float sum = 0.f;
  for (int s = tid; s < S_DIM; s += 256) sum += expf(p[s] - m);
  sdata[tid] = sum; __syncthreads();
  for (int off = 128; off > 0; off >>= 1) {
    if (tid < off) sdata[tid] += sdata[tid + off];
    __syncthreads();
  }
  if (tid == 0) {
    stats[which * 512 + row] = m;
    stats[which * 512 + 256 + row] = logf(sdata[0]);
  }
}

// transposed probabilities: p0t[s,b] = softmax(l0)[b,s]; pst = p0 + p1
__global__ __launch_bounds__(256) void probs_t_kernel(
    const float* __restrict__ l0, const float* __restrict__ l1,
    const float* __restrict__ stats, float* __restrict__ p0t,
    float* __restrict__ pst) {
  int s = blockIdx.x, b = threadIdx.x;
  float p0 = expf(l0[(size_t)b * S_DIM + s] - stats[512 + b] - stats[768 + b]);
  float p1 = expf(l1[(size_t)b * S_DIM + s] - stats[1024 + b] - stats[1280 + b]);
  p0t[(size_t)s * B_DIM + b] = p0;
  pst[(size_t)s * B_DIM + b] = p0 + p1;
}

// dense neighbor weight matrices: Wd[b,nb] += e_k/(2*sum_e), Mw[b,nb] += mask/cnt
__global__ __launch_bounds__(32) void neighbor_weights_kernel(
    const int* __restrict__ neighbors, const float* __restrict__ dists,
    float* __restrict__ Wd, float* __restrict__ Mw) {
  __shared__ float wrow[B_DIM], mrow[B_DIM];
  int b = blockIdx.x, tid = threadIdx.x;
  for (int i = tid; i < B_DIM; i += 32) { wrow[i] = 0.f; mrow[i] = 0.f; }
  __syncthreads();
  if (tid == 0) {
    float se = 0.f, sm = 0.f;
    for (int k = 0; k < K_NB; ++k) {
      float d = dists[b * K_NB + k];
      se += expf(d / 0.6f);
      sm += (d <= 2.0f) ? 1.f : 0.f;
    }
    float cnt = fmaxf(sm, 1.0f);
    for (int k = 0; k < K_NB; ++k) {
      int nb = neighbors[b * K_NB + k];
      float d = dists[b * K_NB + k];
      wrow[nb] += expf(d / 0.6f) / (2.0f * se);
      mrow[nb] += ((d <= 2.0f) ? 1.f : 0.f) / cnt;
    }
  }
  __syncthreads();
  for (int i = tid; i < B_DIM; i += 32) {
    Wd[b * B_DIM + i] = wrow[i];
    Mw[b * B_DIM + i] = mrow[i];
  }
}

// loss_nce, loss_ce, kl -> d_out[0..2] via atomics
__global__ __launch_bounds__(256) void loss_kernel(
    const float* __restrict__ outputs, const float* __restrict__ l0,
    const float* __restrict__ l1, const int* __restrict__ targets,
    const float* __restrict__ stats, const float* __restrict__ LN1,
    const float* __restrict__ LNKL, const float* __restrict__ rampup,
    float* __restrict__ out) {
  __shared__ float sce[256], skl[256];
  int b = blockIdx.x, tid = threadIdx.x;
  float m0 = stats[512 + b], z0 = stats[768 + b];
  float m1 = stats[1024 + b], z1 = stats[1280 + b];
  float ce = 0.f, kl = 0.f;
  for (int s = tid; s < S_DIM; s += 256) {
    float lp0 = l0[(size_t)b * S_DIM + s] - m0 - z0;
    ce += LN1[(size_t)b * S_DIM + s] * lp0;
    float t = LNKL[(size_t)b * S_DIM + s];
    if (t > 0.f) {
      float lp1 = l1[(size_t)b * S_DIM + s] - m1 - z1;
      kl += t * (logf(fmaxf(t, 1e-12f)) - lp1);
    }
  }
  sce[tid] = ce; skl[tid] = kl; __syncthreads();
  for (int off = 128; off > 0; off >>= 1) {
    if (tid < off) { sce[tid] += sce[tid + off]; skl[tid] += skl[tid + off]; }
    __syncthreads();
  }
  if (tid == 0) {
    int t = targets[b];
    float lp0t = l0[(size_t)b * S_DIM + t] - m0 - z0;
    float ce_b = -(0.9f * lp0t + 0.1f * sce[0]);           // ALPHA = 0.9
    float nce_b = -(outputs[(size_t)b * S_DIM + t] - stats[b] - stats[256 + b]);
    atomicAdd(out + 0, nce_b * (1.0f / 256.0f));
    atomicAdd(out + 1, ce_b * (1.0f / 256.0f));            // LAMBDA1 = 1
    atomicAdd(out + 2, rampup[0] * skl[0] * (1.0f / 256.0f));  // LAMBDA2 = 1
  }
}

// new_features = features (scalar copy; d_out+3 is only 4B aligned)
__global__ __launch_bounds__(256) void copy_features_kernel(
    const float* __restrict__ src, float* __restrict__ dst) {
  size_t i = (size_t)blockIdx.x * 256 + threadIdx.x;
  dst[i] = src[i];
}

// scatter momentum update rows: row targets[b] <- normalize(0.2*f + 0.8*x)
__global__ __launch_bounds__(256) void bank_update_kernel(
    const float* __restrict__ features, const float* __restrict__ x,
    const int* __restrict__ targets, float* __restrict__ outF) {
  __shared__ float sred[256];
  int b = blockIdx.x, tid = threadIdx.x;
  int t = targets[b];
  const float* fr = features + (size_t)t * F_DIM;
  const float* xr = x + (size_t)b * F_DIM;
  float u[8], ss = 0.f;
#pragma unroll
  for (int i = 0; i < 8; ++i) {
    int idx = tid + i * 256;
    float v = 0.2f * fr[idx] + 0.8f * xr[idx];   // MOMENTUM = 0.2
    u[i] = v; ss += v * v;
  }
  sred[tid] = ss; __syncthreads();
  for (int off = 128; off > 0; off >>= 1) {
    if (tid < off) sred[tid] += sred[tid + off];
    __syncthreads();
  }
  float inv = 1.0f / fmaxf(sqrtf(sred[0]), 1e-12f);
  float* orow = outF + (size_t)t * F_DIM;
#pragma unroll
  for (int i = 0; i < 8; ++i) orow[tid + i * 256] = u[i] * inv;
}

extern "C" void kernel_launch(void* const* d_in, const int* in_sizes, int n_in,
                              void* d_out, int out_size, void* d_ws, size_t ws_size,
                              hipStream_t stream) {
  const float* inputs0  = (const float*)d_in[0];
  const float* logits0  = (const float*)d_in[1];
  const float* logits1  = (const float*)d_in[2];
  const int*   targets  = (const int*)d_in[3];
  // d_in[4] = indexes (identity, unused)
  const int*   neighbors = (const int*)d_in[5];
  const float* ndists    = (const float*)d_in[6];
  const float* rampup    = (const float*)d_in[7];
  const float* features  = (const float*)d_in[8];
  float* out = (float*)d_out;
  float* ws  = (float*)d_ws;

  // workspace layout (floats, all offsets 256-aligned)
  float* x       = ws;                 // 524288
  float* outputs = ws + 524288;        // 2097152
  float* p0t     = ws + 2621440;       // 2097152 (transposed [S,B])
  float* pst     = ws + 4718592;       // 2097152 (transposed [S,B])
  float* LN1     = ws + 6815744;       // 2097152
  float* LNKL    = ws + 8912896;       // 2097152
  float* Wd      = ws + 11010048;      // 65536
  float* Mw      = ws + 11075584;      // 65536
  float* stats   = ws + 11141120;      // 1536

  normalize_kernel<<<B_DIM, 256, 0, stream>>>(inputs0, x, out);
  // outputs = (x @ features^T) / TEMP
  gemm_nt_bf16x3<<<S_DIM / 32, 64, 0, stream>>>(x, features, outputs,
                                                F_DIM, F_DIM, F_DIM, S_DIM, 20.0f);
  row_stats_kernel<<<3 * B_DIM, 256, 0, stream>>>(outputs, logits0, logits1, stats);
  probs_t_kernel<<<S_DIM, 256, 0, stream>>>(logits0, logits1, stats, p0t, pst);
  neighbor_weights_kernel<<<B_DIM, 32, 0, stream>>>(neighbors, ndists, Wd, Mw);
  // logits_neighbors1 = Wd @ (p0+p1)    (NT against transposed probs)
  gemm_nt_bf16x3<<<S_DIM / 32, 64, 0, stream>>>(Wd, pst, LN1,
                                                B_DIM, B_DIM, B_DIM, S_DIM, 1.0f);
  // logits_neighbors1_KL = Mw @ p0
  gemm_nt_bf16x3<<<S_DIM / 32, 64, 0, stream>>>(Mw, p0t, LNKL,
                                                B_DIM, B_DIM, B_DIM, S_DIM, 1.0f);
  loss_kernel<<<B_DIM, 256, 0, stream>>>(outputs, logits0, logits1, targets,
                                         stats, LN1, LNKL, rampup, out);
  copy_features_kernel<<<(S_DIM * F_DIM) / 256, 256, 0, stream>>>(features, out + 3);
  bank_update_kernel<<<B_DIM, 256, 0, stream>>>(features, x, targets, out + 3);
}
